// IR_Conv_85787676770327
// MI455X (gfx1250) — matrix operand order repaired and checked by reading, not compiled, not run on hardware
//
#include <hip/hip_runtime.h>
#include <hip/hip_bf16.h>
#include <math.h>

#define FL   64          // filter length L
#define NB   128         // batch
#define NT   96000       // time steps (== 375 * 256, so all 256-output tiles are full)
#define TILES_PER_WG 16
#define SEG   (TILES_PER_WG * 256)        // 4096 outputs per workgroup
#define HALO_L 32
#define HALO_R 48
#define XSH_N (SEG + HALO_L + HALO_R)     // 4176 staged x samples
#define NSEG  ((NT + SEG - 1) / SEG)      // 24

typedef __attribute__((ext_vector_type(16))) _Float16 v16h;
typedef __attribute__((ext_vector_type(8)))  float    v8f;

// ---------------------------------------------------------------------------
// Kernel 1: conditioning network -> per-batch 64-tap filter (fp32, tiny)
// ---------------------------------------------------------------------------
__global__ __launch_bounds__(128)
void cond_filter_kernel(const float* __restrict__ c,
                        const float* __restrict__ W_pre,  const float* __restrict__ b_pre,
                        const float* __restrict__ alpha,
                        const float* __restrict__ W_film, const float* __restrict__ b_film,
                        const float* __restrict__ W_glu,  const float* __restrict__ b_glu,
                        const float* __restrict__ W_f,    const float* __restrict__ b_f,
                        float* __restrict__ filt_out) {
    __shared__ float sh_gb[2 * FL];
    __shared__ float sh_h[FL];
    __shared__ float sh_ab[2 * FL];
    __shared__ float sh_g[FL];

    const int b   = blockIdx.x;
    const int tid = threadIdx.x;            // 0..127

    const float c0 = c[b * 3 + 0];
    const float c1 = c[b * 3 + 1];
    const float c2 = c[b * 3 + 2];

    // FiLM params: gb[j] = c . W_film[:,j] + b_film[j]   (j = 0..127)
    float gb = c0 * W_film[0 * 128 + tid] + c1 * W_film[1 * 128 + tid]
             + c2 * W_film[2 * 128 + tid] + b_film[tid];
    sh_gb[tid] = gb;

    // pre-net + leaky relu (j = 0..63)
    if (tid < FL) {
        float pre = c0 * W_pre[0 * 64 + tid] + c1 * W_pre[1 * 64 + tid]
                  + c2 * W_pre[2 * 64 + tid] + b_pre[tid];
        pre = (pre >= 0.0f) ? pre : alpha[tid] * pre;
        __syncthreads();
        sh_h[tid] = pre * sh_gb[tid] + sh_gb[FL + tid];
    } else {
        __syncthreads();
    }
    __syncthreads();

    // GLU input: ab[j] = h . W_glu[:,j] + b_glu[j]   (j = 0..127)
    float ab = b_glu[tid];
    #pragma unroll 8
    for (int i = 0; i < FL; ++i) ab += sh_h[i] * W_glu[i * 128 + tid];
    sh_ab[tid] = ab;
    __syncthreads();

    if (tid < FL) {
        sh_g[tid] = sh_ab[tid] * (1.0f / (1.0f + __expf(-sh_ab[FL + tid])));
    }
    __syncthreads();

    if (tid < FL) {
        float f = b_f[tid];
        #pragma unroll 8
        for (int i = 0; i < FL; ++i) f += sh_g[i] * W_f[i * 64 + tid];
        filt_out[b * FL + tid] = f / (1.0f + fabsf(f));   // softsign
    }
}

// ---------------------------------------------------------------------------
// Kernel 2: per-batch FIR as WMMA matmul against a Toeplitz filter matrix.
//   out[b,t] = sum_k filt[b,k] * x[b, t+k-32],  x zero outside [0,NT)
//   D[m][n] = out[t0+16m+n] = sum_kk A[m][kk]*B[kk][n]
//     A[m][kk] = x[t0-32+16m+kk]   (kk = 0..95, 3 chunks of K=32)
//     B[kk][n] = filt[kk-n] for 0<=kk-n<64 else 0
// ---------------------------------------------------------------------------
__global__ __launch_bounds__(256)
void fir_wmma_kernel(const float* __restrict__ x,
                     const float* __restrict__ filt,
                     float* __restrict__ out) {
    __shared__ __align__(16) _Float16 xsh[XSH_N];
    __shared__ float fsh[FL];

    const int b      = blockIdx.y;
    const int seg    = blockIdx.x;
    const int seg_t0 = seg * SEG;
    const int tid    = threadIdx.x;
    const float* xb  = x + b * NT;          // max offset ~12.3M elements: int is fine

    // filter -> LDS
    if (tid < FL) fsh[tid] = filt[b * FL + tid];

    // stage x[seg_t0-32 .. seg_t0+4144) as f16 into LDS (zero-padded)
    for (int i4 = tid; i4 < XSH_N / 4; i4 += 256) {
        const int t = seg_t0 - HALO_L + 4 * i4;
        float4 v;
        if (t >= 0 && t + 3 < NT) {
            v = *(const float4*)(xb + t);
        } else {
            v.x = (t + 0 >= 0 && t + 0 < NT) ? xb[t + 0] : 0.0f;
            v.y = (t + 1 >= 0 && t + 1 < NT) ? xb[t + 1] : 0.0f;
            v.z = (t + 2 >= 0 && t + 2 < NT) ? xb[t + 2] : 0.0f;
            v.w = (t + 3 >= 0 && t + 3 < NT) ? xb[t + 3] : 0.0f;
        }
        union { _Float16 h[4]; unsigned long long u; } p;
        p.h[0] = (_Float16)v.x; p.h[1] = (_Float16)v.y;
        p.h[2] = (_Float16)v.z; p.h[3] = (_Float16)v.w;
        *(unsigned long long*)&xsh[4 * i4] = p.u;
    }
    __syncthreads();

    const int lane  = tid & 31;
    const int wave  = tid >> 5;
    const int ncol  = lane & 15;                  // B/D column
    const int koff  = (lane < 16) ? 0 : 16;       // B fragment K offset
    const int m     = lane & 15;                  // A row
    const int kbase = (lane < 16) ? 0 : 16;       // A fragment K offset
    const int laneAdd = (lane < 16) ? lane : (112 + lane);  // D store: 16*8+(lane-16)

    // Build the 3 Toeplitz B fragments (K=32 each) in registers.
    // B k order chosen here (same as A): element e <-> K = koff + e (per lane half), col = lane&15.
    v16h bfrag[3];
    #pragma unroll
    for (int cch = 0; cch < 3; ++cch) {
        #pragma unroll
        for (int e = 0; e < 16; ++e) {
            const int k = 32 * cch + koff + e - ncol;
            bfrag[cch][e] = (k >= 0 && k < FL) ? (_Float16)fsh[k] : (_Float16)0.0f;
        }
    }

    // Each wave: 2 tiles of 256 outputs.
    #pragma unroll
    for (int j = 0; j < 2; ++j) {
        const int tl      = 2 * wave + j;                  // 0..15 (wave-uniform)
        const int tile_t0 = seg_t0 + 256 * tl;
        if (tile_t0 >= NT) continue;                       // uniform branch: EXEC stays full
        // NT is a multiple of 256, so tile_t0 < NT implies the whole tile is in
        // range: no per-element store guards needed.

        v8f acc = {};
        const int base = 256 * tl + 16 * m;                // LDS half index (x shift folded in)
        #pragma unroll
        for (int cch = 0; cch < 3; ++cch) {
            union { v16h v; uint4 q[2]; } av;
            av.q[0] = *(const uint4*)(xsh + base + 32 * cch + kbase);       // K kbase..kbase+7
            av.q[1] = *(const uint4*)(xsh + base + 32 * cch + kbase + 8);   // K kbase+8..+15
            acc = __builtin_amdgcn_wmma_f32_16x16x32_f16(
                      false, av.v, false, bfrag[cch], (short)0, acc, false, false);
        }

        float* ob = out + b * NT + tile_t0 + laneAdd;
        #pragma unroll
        for (int v = 0; v < 8; ++v) {
            ob[16 * v] = acc[v];    // 8 coalesced b32 stores, imm offsets 0..448
        }
    }
}

// ---------------------------------------------------------------------------
extern "C" void kernel_launch(void* const* d_in, const int* in_sizes, int n_in,
                              void* d_out, int out_size, void* d_ws, size_t ws_size,
                              hipStream_t stream) {
    const float* inputs = (const float*)d_in[0];
    const float* c      = (const float*)d_in[1];
    const float* W_pre  = (const float*)d_in[2];
    const float* b_pre  = (const float*)d_in[3];
    const float* alpha  = (const float*)d_in[4];
    const float* W_film = (const float*)d_in[5];
    const float* b_film = (const float*)d_in[6];
    const float* W_glu  = (const float*)d_in[7];
    const float* b_glu  = (const float*)d_in[8];
    const float* W_f    = (const float*)d_in[9];
    const float* b_f    = (const float*)d_in[10];

    float* filt = (float*)d_ws;   // 128*64 f32 = 32 KB scratch

    cond_filter_kernel<<<NB, 128, 0, stream>>>(c, W_pre, b_pre, alpha,
                                               W_film, b_film, W_glu, b_glu,
                                               W_f, b_f, filt);

    dim3 grid(NSEG, NB);
    fir_wmma_kernel<<<grid, 256, 0, stream>>>(inputs, filt, (float*)d_out);
}
